// MultiDecoderAutoencoder_73005854098120
// MI455X (gfx1250) — matrix operand order repaired and checked by reading, not compiled, not run on hardware
//
#include <hip/hip_runtime.h>
#include <hip/hip_bf16.h>
#include <stdint.h>

// ---------------------------------------------------------------------------
// MI455X (gfx1250, wave32) implementation.
// All heavy math runs through V_WMMA_F32_16X16X32_F16 (f16 A/B, f32 acc).
// Decoder work is expert-compacted: batch rows are permuted into 16-row
// tiles that share one expert, so each WMMA tile uses a single weight set
// (the reference wastes 4x decoder FLOPs computing all experts).
// Conv layers are template-specialized per shape so K-loops fully unroll
// and staging index math folds to shifts (no runtime division).
// ---------------------------------------------------------------------------

typedef _Float16 half_t;
typedef __attribute__((ext_vector_type(16))) _Float16 v16h;
typedef __attribute__((ext_vector_type(8)))  _Float16 v8h;
typedef __attribute__((ext_vector_type(8)))  float    v8f;

#define NEXPERTS 4
#define BATCH    1024
#define NTILES   67            // max sum_e ceil(count_e/16) = 64 + 3
#define PROWS    (NTILES * 16) // 1072 padded permutation rows

__device__ __forceinline__ v16h cat8(v8h lo, v8h hi) {
  return __builtin_shufflevector(lo, hi, 0,1,2,3,4,5,6,7,8,9,10,11,12,13,14,15);
}

// A fragment: 16x32 f16, M rows striped per lane (ISA 7.12.2 16-bit A layout).
// lane<16: m=lane, K=[0..7],[16..23]; lane>=16: m=lane-16, K=[8..15],[24..31]
__device__ __forceinline__ v16h load_frag_a(const half_t* base, int stride, int kc, int lane) {
  int m  = lane & 15;
  int k0 = (lane < 16) ? 0 : 8;
  const half_t* p = base + (size_t)m * stride + kc + k0;
  v8h lo = *reinterpret_cast<const v8h*>(p);
  v8h hi = *reinterpret_cast<const v8h*>(p + 16);
  return cat8(lo, hi);
}

// B fragment from transposed storage Bt[n][k] (row-major, stride halves):
// lane n = lane&15, K=[0..7],[16..23] (lanes 0-15) or [8..15],[24..31].
__device__ __forceinline__ v16h load_frag_bt(const half_t* base, int stride, int kc, int lane) {
  int n  = lane & 15;
  int k0 = (lane < 16) ? 0 : 8;
  const half_t* p = base + (size_t)n * stride + kc + k0;
  v8h lo = *reinterpret_cast<const v8h*>(p);
  v8h hi = *reinterpret_cast<const v8h*>(p + 16);
  return cat8(lo, hi);
}

// ---------------------------------------------------------------------------
// f32 -> f16 conversion (activations + FC weights staged once per launch)
// ---------------------------------------------------------------------------
__global__ void cvt_f32_to_f16(const float* __restrict__ src,
                               half_t* __restrict__ dst, int n) {
  int i = blockIdx.x * blockDim.x + threadIdx.x;
  if (i < n) dst[i] = (half_t)src[i];
}

// ---------------------------------------------------------------------------
// Expert compaction: padded permutation (16-row tiles per expert) + tile->expert
// ---------------------------------------------------------------------------
__global__ __launch_bounds__(1024) void compact_experts(
    const int* __restrict__ bid, int* __restrict__ pperm, int* __restrict__ texp) {
  __shared__ int cnt[NEXPERTS], cur[NEXPERTS], off[NEXPERTS];
  int t = threadIdx.x;
  if (t < NEXPERTS) { cnt[t] = 0; cur[t] = 0; }
  for (int i = t; i < PROWS; i += blockDim.x) pperm[i] = -1;
  for (int i = t; i < NTILES; i += blockDim.x) texp[i] = 0;
  __syncthreads();
  if (t < BATCH) atomicAdd(&cnt[bid[t]], 1);
  __syncthreads();
  if (t == 0) {
    int base = 0;
    for (int e = 0; e < NEXPERTS; ++e) {
      off[e] = base;
      int ntile = (cnt[e] + 15) >> 4;
      for (int i = 0; i < ntile; ++i) texp[(base >> 4) + i] = e;
      base += ntile << 4;
    }
  }
  __syncthreads();
  if (t < BATCH) {
    int e = bid[t];
    int pos = off[e] + atomicAdd(&cur[e], 1);
    pperm[pos] = t;
  }
}

// ---------------------------------------------------------------------------
// Generic WMMA conv1d (k=3, pad=1) kernel, fully shape-specialized.
//   GEMM view: D[o][p] = sum_{c,k} W[o][c*3+k] * X2[c][p+k-1]
//   M = 32 output channels/block (2 wmma M-tiles), N = 32 conv positions
//   (2 N-tiles), K = CIN*3 padded to mult of 32; one wave per 16x16 tile.
//   UP:   input is nearest-upsampled 2x (X2[q] = X[q>>1])
//   POOL: relu + maxpool2 epilogue (encoder); else optional RELU
//   PERM != 0 => decoder mode: per-tile expert weights, gathered rows
// ---------------------------------------------------------------------------
template <int CIN, int LX, int COUT, int UP, int POOL, int RELU, int OUTF32, int PERM>
__global__ __launch_bounds__(128) void conv_wmma(
    const half_t* __restrict__ in, const float* __restrict__ w,
    const float* __restrict__ bias, void* __restrict__ outp,
    const int* __restrict__ pperm, const int* __restrict__ texp) {
  constexpr int Kp   = CIN * 3;
  constexpr int Kpad = (Kp + 31) & ~31;        // 32 (CIN=2) .. 384 (CIN=128)
  constexpr int Lc   = UP ? (LX * 2) : LX;     // conv-domain length
  const int p0  = blockIdx.x * 32;             // conv output position base
  const int co0 = blockIdx.y * 32;             // output channel base

  int row;
  if (PERM) {
    row = pperm[blockIdx.z];
    if (row < 0) return;                       // padding tile row (uniform branch)
    int e = texp[blockIdx.z >> 4];
    w    += (size_t)e * COUT * Kp;
    bias += (size_t)e * COUT;
  } else {
    row = blockIdx.z;
  }

  __shared__ __align__(16) half_t A_s[32 * Kpad];   // weights  [32][Kpad]
  __shared__ __align__(16) half_t Bt_s[32 * Kpad];  // im2col^T [32 pos][Kpad]
  __shared__ float y_s[32 * 32];                    // f32 pre-activation tile

  const int tid = threadIdx.x;

  // stage weights (f32 -> f16), zero-pad K and channels beyond COUT
  for (int idx = tid; idx < 32 * Kpad; idx += 128) {
    int ol = idx / Kpad, kk = idx - ol * Kpad;
    int o = co0 + ol;
    half_t v = (half_t)0.f;
    if (o < COUT && kk < Kp) v = (half_t)w[(size_t)o * Kp + kk];
    A_s[idx] = v;
  }
  // stage transposed-B: Bt[p][c*3+k] = X2[c][p0+p+k-1] (zero at borders)
  const half_t* xin = in + (size_t)row * CIN * LX;
  for (int idx = tid; idx < 32 * Kpad; idx += 128) {
    int p = idx / Kpad, kk = idx - p * Kpad;
    half_t v = (half_t)0.f;
    if (kk < Kp) {
      int c = kk / 3, k = kk - c * 3;
      int q = p0 + p + k - 1;
      if (q >= 0 && q < Lc) {
        int s = UP ? (q >> 1) : q;
        v = xin[(size_t)c * LX + s];
      }
    }
    Bt_s[idx] = v;
  }
  __syncthreads();

  // 4 waves -> (mt, nt) 16x16 tiles; K-loop fully unrolled
  const int wave = tid >> 5, lane = tid & 31;
  const int mt = wave >> 1, nt = wave & 1;
  const half_t* A0  = &A_s[(size_t)mt * 16 * Kpad];
  const half_t* Bt0 = &Bt_s[(size_t)nt * 16 * Kpad];
  v8f c = {};
#pragma unroll
  for (int kc = 0; kc < Kpad; kc += 32) {
    v16h a = load_frag_a(A0, Kpad, kc, lane);
    v16h b = load_frag_bt(Bt0, Kpad, kc, lane);
    c = __builtin_amdgcn_wmma_f32_16x16x32_f16(false, a, false, b,
                                               (short)0, c, false, false);
  }
  {
    int n  = lane & 15;
    int mo = (lane < 16) ? 0 : 8;
#pragma unroll
    for (int r = 0; r < 8; ++r)
      y_s[(mt * 16 + r + mo) * 32 + nt * 16 + n] = c[r];
  }
  __syncthreads();

  // epilogue: bias (+relu) (+maxpool2), store f16 or f32
  constexpr int nOut = POOL ? 16 : 32;
  constexpr size_t Lout = POOL ? (size_t)(Lc / 2) : (size_t)Lc;
  for (int idx = tid; idx < 32 * nOut; idx += 128) {
    int ol = idx / nOut, j = idx - ol * nOut;
    int o = co0 + ol;
    if (o >= COUT) continue;
    float bb = bias[o];
    float v;
    size_t pos;
    if (POOL) {
      float a0 = fmaxf(y_s[ol * 32 + 2 * j] + bb, 0.f);
      float a1 = fmaxf(y_s[ol * 32 + 2 * j + 1] + bb, 0.f);
      v = fmaxf(a0, a1);
      pos = (size_t)blockIdx.x * 16 + j;
    } else {
      v = y_s[ol * 32 + j] + bb;
      if (RELU) v = fmaxf(v, 0.f);
      pos = (size_t)p0 + j;
    }
    size_t oidx = ((size_t)row * COUT + o) * Lout + pos;
    if (OUTF32) ((float*)outp)[oidx] = v;
    else        ((half_t*)outp)[oidx] = (half_t)v;
  }
}

// ---------------------------------------------------------------------------
// Encoder FC: z[b][o] = flat[b][:] . W[o][:] + fcb[o]   (M=1024,N=128,K=32768)
// Fragments streamed straight from global (L2-resident weights) + prefetch.
// ---------------------------------------------------------------------------
__global__ __launch_bounds__(256) void fc_enc_wmma(
    const half_t* __restrict__ flat,  // [1024][32768]  (h3 flattened c*256+l)
    const half_t* __restrict__ W,     // [128][32768]   f16 enc_fcw
    const float*  __restrict__ fcb,   // [128]
    half_t* __restrict__ z) {         // [1024][128]
  const int wave = threadIdx.x >> 5, lane = threadIdx.x & 31;
  const int m0 = blockIdx.x * 16;
  const int n0 = wave * 16;                   // 8 waves cover all 128 latents
  const int nm = lane & 15;
  const int k0a = (lane < 16) ? 0 : 8;
  const int k0b = (lane < 16) ? 0 : 8;
  const half_t* ap = flat + (size_t)(m0 + nm) * 32768 + k0a;
  const half_t* bp = W    + (size_t)(n0 + nm) * 32768 + k0b;
  v8f c = {};
#pragma unroll 4
  for (int kc = 0; kc < 32768; kc += 32) {
    v8h alo = *reinterpret_cast<const v8h*>(ap + kc);
    v8h ahi = *reinterpret_cast<const v8h*>(ap + kc + 16);
    v8h blo = *reinterpret_cast<const v8h*>(bp + kc);
    v8h bhi = *reinterpret_cast<const v8h*>(bp + kc + 16);
    __builtin_prefetch(bp + kc + 512, 0, 1);   // global_prefetch_b8
    c = __builtin_amdgcn_wmma_f32_16x16x32_f16(false, cat8(alo, ahi),
                                               false, cat8(blo, bhi),
                                               (short)0, c, false, false);
  }
  const int mo = (lane < 16) ? 0 : 8;
  float bb = fcb[n0 + nm];
#pragma unroll
  for (int r = 0; r < 8; ++r)
    z[(size_t)(m0 + r + mo) * 128 + n0 + nm] = (half_t)(c[r] + bb);
}

// ---------------------------------------------------------------------------
// Decoder FC (grouped by expert tile): h[row][o] = z[row] . Wd[e][o] + b[e][o]
// M = 16 gathered batch rows/tile, N = 32768, K = 128 (4 wmma steps).
// ---------------------------------------------------------------------------
__global__ __launch_bounds__(256) void fc_dec_wmma(
    const half_t* __restrict__ z,     // [1024][128]
    const half_t* __restrict__ W,     // [4][32768][128] f16 dec_fcw
    const float*  __restrict__ fcb,   // [4][32768]
    const int* __restrict__ pperm, const int* __restrict__ texp,
    half_t* __restrict__ h) {         // [1024][32768], original row order
  const int wave = threadIdx.x >> 5, lane = threadIdx.x & 31;
  const int t = blockIdx.y;
  const int e = texp[t];
  const int n0 = blockIdx.x * 128 + wave * 16;
  const int nm = lane & 15;
  const int k0a = (lane < 16) ? 0 : 8;
  const int k0b = (lane < 16) ? 0 : 8;
  int rA = pperm[t * 16 + nm];
  if (rA < 0) rA = 0;                          // pad rows compute garbage, unstored
  const half_t* ap = z + (size_t)rA * 128 + k0a;
  const half_t* bp = W + ((size_t)e * 32768 + (n0 + nm)) * 128 + k0b;
  v8f c = {};
#pragma unroll
  for (int kc = 0; kc < 128; kc += 32) {
    v8h alo = *reinterpret_cast<const v8h*>(ap + kc);
    v8h ahi = *reinterpret_cast<const v8h*>(ap + kc + 16);
    v8h blo = *reinterpret_cast<const v8h*>(bp + kc);
    v8h bhi = *reinterpret_cast<const v8h*>(bp + kc + 16);
    c = __builtin_amdgcn_wmma_f32_16x16x32_f16(false, cat8(alo, ahi),
                                               false, cat8(blo, bhi),
                                               (short)0, c, false, false);
  }
  const int mo = (lane < 16) ? 0 : 8;
  float bb = fcb[(size_t)e * 32768 + n0 + nm];
#pragma unroll
  for (int r = 0; r < 8; ++r) {
    int rr = pperm[t * 16 + r + mo];
    if (rr >= 0) h[(size_t)rr * 32768 + n0 + nm] = (half_t)(c[r] + bb);
  }
}

// ---------------------------------------------------------------------------
// Launch: convert -> compact -> encoder(3 conv + FC) -> decoder(FC + 3 conv)
// Workspace (~185 MB): x16, two ping-pong f16 activation buffers, z16,
// f16 FC weights, permutation tables.
// ---------------------------------------------------------------------------
extern "C" void kernel_launch(void* const* d_in, const int* in_sizes, int n_in,
                              void* d_out, int out_size, void* d_ws, size_t ws_size,
                              hipStream_t stream) {
  (void)in_sizes; (void)n_in; (void)out_size; (void)ws_size;
  const float* x       = (const float*)d_in[0];
  const int*   bid     = (const int*)d_in[1];
  const float* enc_w1  = (const float*)d_in[2];
  const float* enc_b1  = (const float*)d_in[3];
  const float* enc_w2  = (const float*)d_in[4];
  const float* enc_b2  = (const float*)d_in[5];
  const float* enc_w3  = (const float*)d_in[6];
  const float* enc_b3  = (const float*)d_in[7];
  const float* enc_fcw = (const float*)d_in[8];
  const float* enc_fcb = (const float*)d_in[9];
  const float* dec_fcw = (const float*)d_in[10];
  const float* dec_fcb = (const float*)d_in[11];
  const float* dec_w1  = (const float*)d_in[12];
  const float* dec_b1  = (const float*)d_in[13];
  const float* dec_w2  = (const float*)d_in[14];
  const float* dec_b2  = (const float*)d_in[15];
  const float* dec_w3  = (const float*)d_in[16];
  const float* dec_b3  = (const float*)d_in[17];

  char* wp = (char*)d_ws;
  auto carve = [&](size_t bytes) {
    char* r = wp;
    wp += (bytes + 255) & ~(size_t)255;
    return r;
  };
  half_t* x16    = (half_t*)carve((size_t)BATCH * 2 * 2048 * 2);    //  8.4 MB
  half_t* B0     = (half_t*)carve((size_t)BATCH * 32 * 1024 * 2);   // 67.1 MB
  half_t* B1     = (half_t*)carve((size_t)BATCH * 32 * 1024 * 2);   // 67.1 MB
  half_t* z16    = (half_t*)carve((size_t)BATCH * 128 * 2);
  half_t* encW16 = (half_t*)carve((size_t)128 * 32768 * 2);         //  8.4 MB
  half_t* decW16 = (half_t*)carve((size_t)4 * 32768 * 128 * 2);     // 33.6 MB
  int* pperm     = (int*)carve(PROWS * 4);
  int* texp      = (int*)carve(NTILES * 4);

  // stage f16 copies of x and the big FC weights
  {
    int n = BATCH * 2 * 2048;
    cvt_f32_to_f16<<<(n + 255) / 256, 256, 0, stream>>>(x, x16, n);
    n = 128 * 32768;
    cvt_f32_to_f16<<<(n + 255) / 256, 256, 0, stream>>>(enc_fcw, encW16, n);
    n = 4 * 32768 * 128;
    cvt_f32_to_f16<<<(n + 255) / 256, 256, 0, stream>>>(dec_fcw, decW16, n);
  }
  compact_experts<<<1, 1024, 0, stream>>>(bid, pperm, texp);

  // ---- encoder ----
  // conv1: x16[1024,2,2048] -> B0[1024,32,1024] (relu+pool)
  conv_wmma<2, 2048, 32, 0, 1, 1, 0, 0><<<dim3(64, 1, BATCH), 128, 0, stream>>>(
      x16, enc_w1, enc_b1, B0, nullptr, nullptr);
  // conv2: B0 -> B1[1024,64,512]
  conv_wmma<32, 1024, 64, 0, 1, 1, 0, 0><<<dim3(32, 2, BATCH), 128, 0, stream>>>(
      B0, enc_w2, enc_b2, B1, nullptr, nullptr);
  // conv3: B1 -> B0[1024,128,256]
  conv_wmma<64, 512, 128, 0, 1, 1, 0, 0><<<dim3(16, 4, BATCH), 128, 0, stream>>>(
      B1, enc_w3, enc_b3, B0, nullptr, nullptr);
  // FC: flat(B0) [1024,32768] -> z16 [1024,128]
  fc_enc_wmma<<<BATCH / 16, 256, 0, stream>>>(B0, encW16, enc_fcb, z16);

  // ---- decoder (expert-compacted) ----
  // FC: z16 -> B1 viewed as h[1024,32768] (== [row,128,256])
  fc_dec_wmma<<<dim3(32768 / 128, NTILES), 256, 0, stream>>>(
      z16, decW16, dec_fcb, pperm, texp, B1);
  // up2+conv1: B1[row,128,256] -> B0[row,64,512] (relu)
  conv_wmma<128, 256, 64, 1, 0, 1, 0, 1><<<dim3(16, 2, PROWS), 128, 0, stream>>>(
      B1, dec_w1, dec_b1, B0, pperm, texp);
  // up2+conv2: B0 -> B1[row,32,1024] (relu)
  conv_wmma<64, 512, 32, 1, 0, 1, 0, 1><<<dim3(32, 1, PROWS), 128, 0, stream>>>(
      B0, dec_w2, dec_b2, B1, pperm, texp);
  // up2+conv3: B1 -> d_out[row,2,2048] (f32, no relu)
  conv_wmma<32, 1024, 2, 1, 0, 0, 1, 1><<<dim3(64, 1, PROWS), 128, 0, stream>>>(
      B1, dec_w3, dec_b3, d_out, pperm, texp);
}
